// SingleHeadAttentionLayer_6811818131666
// MI455X (gfx1250) — matrix operand order repaired and checked by reading, not compiled, not run on hardware
//
#include <hip/hip_runtime.h>

typedef __attribute__((ext_vector_type(16))) _Float16 v16h;
typedef __attribute__((ext_vector_type(8)))  _Float16 v8h;
typedef __attribute__((ext_vector_type(8)))  float    v8f;

#define LQ  8192
#define LKK 8192
#define DIN 128   // Q_SIZE = K_SIZE
#define DA  64    // A_SIZE (attention dim)
#define DV  64    // V_SIZE (output dim)

static __device__ __forceinline__ v8f wmma_f16(v16h a, v16h b, v8f c) {
  // (neg_a, A, neg_b, B, c_mod, C, reuse_a, reuse_b)
  return __builtin_amdgcn_wmma_f32_16x16x32_f16(false, a, false, b, (short)0, c,
                                                false, false);
}

// Low 32 bits of a generic pointer to __shared__ = allocation-relative LDS
// byte offset (local->flat addrspacecast is {aperture_hi, lds_offset_lo}).
static __device__ __forceinline__ unsigned lds_lo32(const void* p) {
  return (unsigned)(unsigned long long)(uintptr_t)p;
}

// One wave-wide async DMA: each lane moves 16B global -> LDS. ASYNCcnt-tracked.
static __device__ __forceinline__ void async_copy_b128(unsigned lds_byte_off,
                                                       const void* gptr) {
  asm volatile("global_load_async_to_lds_b128 %0, %1, off"
               :: "v"(lds_byte_off),
                  "v"((unsigned long long)(uintptr_t)gptr)
               : "memory");
}

static __device__ __forceinline__ void wait_async0() {
  asm volatile("s_wait_asynccnt 0x0" ::: "memory");
}

// ---------------------------------------------------------------------------
// Kernel 1: fused projections  X[8192,128] @ W[128,64] + b  ->  f16 in d_ws
// blockIdx.y: 0=Q (scale 1/8 folded in), 1=K, 2=V stored TRANSPOSED [64][8192]
// so the attention kernel can stage V with contiguous 16B async copies.
// ---------------------------------------------------------------------------
__global__ __launch_bounds__(128) void proj_kernel(
    const float* __restrict__ q, const float* __restrict__ k,
    const float* __restrict__ v,
    const float* __restrict__ Wq, const float* __restrict__ bq,
    const float* __restrict__ Wk, const float* __restrict__ bk,
    const float* __restrict__ Wv, const float* __restrict__ bv,
    _Float16* __restrict__ Qp, _Float16* __restrict__ Kp,
    _Float16* __restrict__ VpT)
{
  const int lane = threadIdx.x & 31;
  const int wave = threadIdx.x >> 5;        // N tile 0..3
  const int ln   = lane & 15;
  const int kb8  = (lane >> 4) * 8;         // unused; A-frag also uses kb16
  const int kb16 = (lane >> 4) * 16;        // B-frag K sub-base
  const int mb   = (lane >> 4) * 8;         // C-frag M base
  const int row0 = blockIdx.x * 16;

  const float* src; const float* W; const float* bias; _Float16* dst;
  float scale; bool transposed;
  if (blockIdx.y == 0)      { src = q; W = Wq; bias = bq; dst = Qp;  scale = 0.125f; transposed = false; }
  else if (blockIdx.y == 1) { src = k; W = Wk; bias = bk; dst = Kp;  scale = 1.0f;   transposed = false; }
  else                      { src = v; W = Wv; bias = bv; dst = VpT; scale = 1.0f;   transposed = true;  }

  const int col = wave * 16 + ln;
  v8f acc = {};
  #pragma unroll
  for (int kk = 0; kk < DIN; kk += 32) {
    v16h a;
    #pragma unroll
    for (int i = 0; i < 16; ++i) {
      const int klocal = kb16 + i;
      a[i] = (_Float16)src[(row0 + ln) * DIN + kk + klocal];
    }
    v16h b;
    #pragma unroll
    for (int i = 0; i < 16; ++i)
      b[i] = (_Float16)W[(kk + kb16 + i) * DA + col];
    acc = wmma_f16(a, b, acc);
  }
  const float bc = bias[col];
  #pragma unroll
  for (int r = 0; r < 8; ++r) {
    const float val = (acc[r] + bc) * scale;
    if (transposed)
      dst[(size_t)col * LKK + (row0 + mb + r)] = (_Float16)val;  // 8 consecutive f16/lane
    else
      dst[(row0 + mb + r) * DA + col] = (_Float16)val;
  }
}

// Issue one 32-key chunk's staging: 4 async b128 per wave (K 4KB + Vt 4KB).
static __device__ __forceinline__ void stage_chunk(
    const _Float16* __restrict__ Kp, const _Float16* __restrict__ VpT,
    int j, int tid, unsigned klds, unsigned vlds)
{
  const char* kg = (const char*)(Kp + (size_t)j * DA);   // contiguous 4KB
  async_copy_b128(klds + (unsigned)(tid * 16),         kg + tid * 16);
  async_copy_b128(klds + (unsigned)((tid + 128) * 16), kg + (tid + 128) * 16);
  // Vt chunk: column c occupies 32 f16 = 64B at VpT + c*LKK + j.
  const int u0 = tid, u1 = tid + 128;                    // 16B units 0..255
  const int c0 = u0 >> 2, s0 = (u0 & 3) * 8;             // s in f16 elements
  const int c1 = u1 >> 2, s1 = (u1 & 3) * 8;
  async_copy_b128(vlds + (unsigned)((c0 * 32 + s0) * 2),
                  (const char*)(VpT + (size_t)c0 * LKK + j + s0));
  async_copy_b128(vlds + (unsigned)((c1 * 32 + s1) * 2),
                  (const char*)(VpT + (size_t)c1 * LKK + j + s1));
}

// ---------------------------------------------------------------------------
// Kernel 2: flash attention. 128 blocks x 128 threads (4 waves), wave owns 16
// query rows. Double-buffered async-to-LDS staging of 32-key K/V chunks.
// Per chunk: 4 WMMA (S), online softmax, P bounce through LDS, 4 WMMA (O).
// ---------------------------------------------------------------------------
__global__ __launch_bounds__(128) void attn_kernel(
    const _Float16* __restrict__ Qp, const _Float16* __restrict__ Kp,
    const _Float16* __restrict__ VpT, float* __restrict__ out)
{
  __shared__ __align__(32) _Float16 sK [2][32 * 64];   // K chunk [key][kdim]
  __shared__ __align__(32) _Float16 sVt[2][64 * 32];   // V chunk [col][key]
  __shared__ __align__(32) _Float16 sP [4 * 16 * 32];  // per-wave exp(S) tiles

  const int tid  = threadIdx.x;
  const int lane = tid & 31;
  const int wave = tid >> 5;
  const int ln   = lane & 15;
  const int kb8  = (lane >> 4) * 8;
  const int kb16 = (lane >> 4) * 16;
  const int mb   = (lane >> 4) * 8;

  const int qrow = blockIdx.x * 64 + wave * 16;

  // Q A-frags (K dim 0..31 and 32..63); 1/sqrt(dA) folded into Qp.
  v16h aq0, aq1;
  #pragma unroll
  for (int i = 0; i < 16; ++i) {
    const int klocal = kb16 + i;
    aq0[i] = Qp[(qrow + ln) * DA + klocal];
    aq1[i] = Qp[(qrow + ln) * DA + 32 + klocal];
  }

  const unsigned klds[2] = { lds_lo32(&sK[0][0]),  lds_lo32(&sK[1][0]) };
  const unsigned vlds[2] = { lds_lo32(&sVt[0][0]), lds_lo32(&sVt[1][0]) };

  float mrun[8], lrun[8];
  v8f o0 = {}, o1 = {}, o2 = {}, o3 = {};
  #pragma unroll
  for (int r = 0; r < 8; ++r) { mrun[r] = -1e30f; lrun[r] = 0.0f; }

  _Float16* sPw = sP + wave * (16 * 32);

  // Prologue: start DMA of chunk 0 into buffer 0.
  stage_chunk(Kp, VpT, 0, tid, klds[0], vlds[0]);
  int cur = 0;

  for (int j = 0; j < LKK; j += 32) {
    wait_async0();        // my async stores into buf[cur] have landed in LDS
    __syncthreads();      // everyone's landed; prior reads of buf[cur^1] done
    if (j + 32 < LKK)     // overlap next chunk's DMA with this chunk's math
      stage_chunk(Kp, VpT, j + 32, tid, klds[cur ^ 1], vlds[cur ^ 1]);

    const _Float16* K  = sK[cur];
    const _Float16* Vt = sVt[cur];

    // S tile: 16 rows x 32 keys; K-dim 64 in 2 WMMA steps per key tile.
    v8f c0 = {}, c1 = {};
    {
      v16h b;
      b = *(const v16h*)&K[ ln       * DA +  0 + kb16]; c0 = wmma_f16(aq0, b, c0);
      b = *(const v16h*)&K[ ln       * DA + 32 + kb16]; c0 = wmma_f16(aq1, b, c0);
      b = *(const v16h*)&K[(16 + ln) * DA +  0 + kb16]; c1 = wmma_f16(aq0, b, c1);
      b = *(const v16h*)&K[(16 + ln) * DA + 32 + kb16]; c1 = wmma_f16(aq1, b, c1);
    }

    // Online softmax (row reductions across 16-lane halves).
    float corr[8];
    #pragma unroll
    for (int r = 0; r < 8; ++r) {
      float cm = fmaxf(c0[r], c1[r]);
      cm = fmaxf(cm, __shfl_xor(cm, 1));
      cm = fmaxf(cm, __shfl_xor(cm, 2));
      cm = fmaxf(cm, __shfl_xor(cm, 4));
      cm = fmaxf(cm, __shfl_xor(cm, 8));
      const float mnew = fmaxf(mrun[r], cm);
      const float cr = __expf(mrun[r] - mnew);
      const float p0 = __expf(c0[r] - mnew);
      const float p1 = __expf(c1[r] - mnew);
      float rs = p0 + p1;
      rs += __shfl_xor(rs, 1);
      rs += __shfl_xor(rs, 2);
      rs += __shfl_xor(rs, 4);
      rs += __shfl_xor(rs, 8);
      lrun[r] = lrun[r] * cr + rs;
      mrun[r] = mnew;
      corr[r] = cr;
      sPw[(mb + r) * 32 + ln]      = (_Float16)p0;   // C-layout -> LDS
      sPw[(mb + r) * 32 + 16 + ln] = (_Float16)p1;
    }
    #pragma unroll
    for (int r = 0; r < 8; ++r) {
      o0[r] *= corr[r]; o1[r] *= corr[r]; o2[r] *= corr[r]; o3[r] *= corr[r];
    }

    // Wave-local LDS RAW on sP: lockstep wave32 + DS-counter wait suffices.
    asm volatile("s_wait_dscnt 0" ::: "memory");

    // P A-frag (16x32) gathered in A layout from the per-wave LDS tile.
    v8h plo = *(const v8h*)&sPw[ln * 32 + kb16];
    v8h phi = *(const v8h*)&sPw[ln * 32 + kb16 + 8];
    v16h ap = __builtin_shufflevector(plo, phi,
                                      0, 1, 2, 3, 4, 5, 6, 7,
                                      8, 9, 10, 11, 12, 13, 14, 15);

    // O += P @ V : 4 output column tiles, contiguous B-frags from Vt.
    v16h b0 = *(const v16h*)&Vt[(0 * 16 + ln) * 32 + kb16];
    v16h b1 = *(const v16h*)&Vt[(1 * 16 + ln) * 32 + kb16];
    v16h b2 = *(const v16h*)&Vt[(2 * 16 + ln) * 32 + kb16];
    v16h b3 = *(const v16h*)&Vt[(3 * 16 + ln) * 32 + kb16];
    o0 = wmma_f16(ap, b0, o0);
    o1 = wmma_f16(ap, b1, o1);
    o2 = wmma_f16(ap, b2, o2);
    o3 = wmma_f16(ap, b3, o3);

    cur ^= 1;
  }

  #pragma unroll
  for (int r = 0; r < 8; ++r) {
    const float inv = 1.0f / lrun[r];
    const int row = qrow + mb + r;
    out[row * DV +  0 + ln] = o0[r] * inv;
    out[row * DV + 16 + ln] = o1[r] * inv;
    out[row * DV + 32 + ln] = o2[r] * inv;
    out[row * DV + 48 + ln] = o3[r] * inv;
  }
}

// ---------------------------------------------------------------------------
extern "C" void kernel_launch(void* const* d_in, const int* in_sizes, int n_in,
                              void* d_out, int out_size, void* d_ws, size_t ws_size,
                              hipStream_t stream) {
  (void)in_sizes; (void)n_in; (void)out_size; (void)ws_size;
  const float* q  = (const float*)d_in[0];
  const float* k  = (const float*)d_in[1];
  const float* v  = (const float*)d_in[2];
  const float* Wq = (const float*)d_in[3];
  const float* bq = (const float*)d_in[4];
  const float* Wk = (const float*)d_in[5];
  const float* bk = (const float*)d_in[6];
  const float* Wv = (const float*)d_in[7];
  const float* bv = (const float*)d_in[8];

  _Float16* Qp  = (_Float16*)d_ws;           // [8192][64] f16, 1 MB
  _Float16* Kp  = Qp + (size_t)LQ * DA;      // [8192][64] f16, 1 MB
  _Float16* VpT = Kp + (size_t)LKK * DA;     // [64][8192] f16 (transposed), 1 MB

  proj_kernel<<<dim3(LQ / 16, 3), 128, 0, stream>>>(q, k, v, Wq, bq, Wk, bk,
                                                    Wv, bv, Qp, Kp, VpT);
  attn_kernel<<<dim3(LQ / 64), 128, 0, stream>>>(Qp, Kp, VpT, (float*)d_out);
}
